// VectorQuantizer_17437567221786
// MI455X (gfx1250) — matrix operand order repaired and checked by reading, not compiled, not run on hardware
//
#include <hip/hip_runtime.h>

typedef __attribute__((ext_vector_type(16))) __bf16 v16bf;
typedef __attribute__((ext_vector_type(8)))  float  v8f;

#define KCB   1024      // codebook entries
#define DIM   64        // embedding dim
#define TT    4096      // T
#define BB    16        // batch
#define NROWS (BB * TT) // 65536 vectors
#define ROWS_PER_BLOCK 256
#define NBLK  (NROWS / ROWS_PER_BLOCK)   // 256
#define KPHASE 256      // codebook rows staged in LDS per phase
#define NPHASE (KCB / KPHASE)            // 4
#define WSTR  68        // LDS codebook row stride (bf16 elems) = 136B (2-way conflicts max)
#define XSTR  72        // LDS latent row stride (bf16 elems) = 144B
#define BETA  0.98f

#if defined(__has_builtin)
#  if __has_builtin(__builtin_amdgcn_tensor_load_to_lds)
#    define HAS_TDM 1
#  endif
#endif
#ifndef HAS_TDM
#  define HAS_TDM 0
#endif

union AFrag {
    v16bf  v;
    float4 f4[2];
    float2 f2[4];
};

__device__ __forceinline__ v8f wmma_bf16(const AFrag& a, const AFrag& b, v8f c) {
    return __builtin_amdgcn_wmma_f32_16x16x32_bf16(
        false, a.v, false, b.v, (short)0, c, false, false);
}

#if HAS_TDM
typedef __attribute__((ext_vector_type(4))) unsigned int v4u;
typedef __attribute__((ext_vector_type(8))) int          v8i_;
typedef __attribute__((ext_vector_type(4))) int          v4i_;

// 2-D TDM load: `rows` x 64 bf16 tile, row-major in global, padded rows in LDS
// (pad 2 DWORDs after every 32 DWORDs => 136B LDS row stride).
__device__ __forceinline__ void tdm_load_2d(unsigned lds_off, const void* gptr, int rows) {
    unsigned long long ga = (unsigned long long)gptr;
    v4u g0;
    g0[0] = 1u;                                   // count=1, user descriptor
    g0[1] = lds_off;                              // LDS byte address
    g0[2] = (unsigned)(ga & 0xFFFFFFFFu);         // global_addr[31:0]
    g0[3] = (unsigned)((ga >> 32) & 0x01FFFFFFu)  // global_addr[56:32]
          | (2u << 30);                           // type = 2 ("image")
    v8i_ g1;
    g1[0] = (1 << 16)                             // data_size = 2 bytes
          | (1 << 20)                             // pad_enable
          | (4 << 22)                             // pad_interval: 32 DWORDs (one row)
          | (1 << 25);                            // pad_amount: 2 DWORDs (4 bf16)
    g1[1] = (DIM << 16);                          // tensor_dim0 = 64 (bits 79:48 lo)
    g1[2] = (rows & 0xFFFF) << 16;                // tensor_dim1 lo16 (bits 111:80)
    g1[3] = (DIM << 16);                          // dim1 hi16 = 0 | tile_dim0 = 64
    g1[4] = rows;                                 // tile_dim1 | tile_dim2 = 0
    g1[5] = DIM;                                  // tensor_dim0_stride = 64 (lo32)
    g1[6] = 0;                                    // stride0 hi | dim1_stride lo = 0
    g1[7] = 0;
    v4i_ z4 = {0, 0, 0, 0};
    v8i_ z8 = {0, 0, 0, 0, 0, 0, 0, 0};
    // clang-23 / therock-10.0 lane: 6-arg form (g0, g1, g2, g3, extra, cpol)
    __builtin_amdgcn_tensor_load_to_lds(g0, g1, z4, z4, z8, 0);
}
#endif

// ---------------------------------------------------------------------------
// Kernel 0: split codebook into bf16 hi/lo, compute ||w_k||^2
// ---------------------------------------------------------------------------
__global__ __launch_bounds__(256) void vq_prep(const float* __restrict__ W,
                                               unsigned short* __restrict__ whi_u,
                                               unsigned short* __restrict__ wlo_u,
                                               float* __restrict__ wnorm) {
    int k = blockIdx.x * 256 + threadIdx.x;
    if (k >= KCB) return;
    __bf16* whi = (__bf16*)whi_u;
    __bf16* wlo = (__bf16*)wlo_u;
    float acc = 0.f;
#pragma unroll 8
    for (int d = 0; d < DIM; ++d) {
        float w = W[k * DIM + d];
        __bf16 h = (__bf16)w;
        __bf16 l = (__bf16)(w - (float)h);
        whi[k * DIM + d] = h;
        wlo[k * DIM + d] = l;
        acc += w * w;
    }
    wnorm[k] = acc;
}

// ---------------------------------------------------------------------------
// Kernel 1: 256 rows/block, 2 row-tiles per wave, codebook phase-staged in LDS
// via the Tensor Data Mover. argmin_k(||w||^2 - 2 x.w) with bf16 hi/lo WMMA.
// ---------------------------------------------------------------------------
__global__ __launch_bounds__(256) void vq_main(const float* __restrict__ lat,
                                               const float* __restrict__ W,
                                               const unsigned short* __restrict__ whi_u,
                                               const unsigned short* __restrict__ wlo_u,
                                               const float* __restrict__ wnorm,
                                               float* __restrict__ out,
                                               float* __restrict__ partials) {
    __shared__ __bf16 Wh_lds[KPHASE * WSTR];     // 34 KB
    __shared__ __bf16 Wl_lds[KPHASE * WSTR];     // 34 KB
    __shared__ __bf16 xh[ROWS_PER_BLOCK * XSTR]; // 36 KB
    __shared__ __bf16 xl[ROWS_PER_BLOCK * XSTR]; // 36 KB
    __shared__ float  wn_lds[KCB];
    __shared__ int    ind_lds[ROWS_PER_BLOCK];
    __shared__ float  red[8];

    const int tid = threadIdx.x;
    const int blk = blockIdx.x;
    const int rowBase = blk * ROWS_PER_BLOCK;   // all rows in same b
    const int b  = rowBase >> 12;
    const int t0 = rowBase & (TT - 1);
    const float* latBase = lat + (size_t)b * DIM * TT + t0;
    float*       outBase = out + (size_t)b * DIM * TT + t0;

    const int wave = tid >> 5;
    const int lane = tid & 31;
    const int n    = lane & 15;   // A row / B,C column slot
    const int kg   = lane >> 4;

#if HAS_TDM
    if (wave == 0) {   // kick phase-0 codebook DMA before latents staging
        tdm_load_2d((unsigned)(size_t)(void*)Wh_lds, whi_u, KPHASE);
        tdm_load_2d((unsigned)(size_t)(void*)Wl_lds, wlo_u, KPHASE);
    }
#endif

    // ---- stage latents tile [256 rows][64 d] as bf16 hi/lo ----
    {
        const int r = tid;
        for (int d = 0; d < DIM; ++d) {
            float v = latBase[(size_t)d * TT + r];
            __bf16 h = (__bf16)v;
            __bf16 l = (__bf16)(v - (float)h);
            xh[r * XSTR + d] = h;
            xl[r * XSTR + d] = l;
        }
    }
    for (int k = tid; k < KCB; k += 256) wn_lds[k] = wnorm[k];
#if !HAS_TDM
    for (int i = tid; i < KPHASE * DIM; i += 256) {
        int row = i >> 6, d = i & 63;
        Wh_lds[row * WSTR + d] = ((const __bf16*)whi_u)[i];
        Wl_lds[row * WSTR + d] = ((const __bf16*)wlo_u)[i];
    }
#endif
    __syncthreads();

    // ---- A fragments for both tiles (ISA 16-bit A 16x32 layout) ----
    AFrag ah0[2], ah1[2], al0[2], al1[2];
#pragma unroll
    for (int t = 0; t < 2; ++t) {
        const int lr = wave * 32 + t * 16 + n;
        const __bf16* bh = &xh[lr * XSTR];
        const __bf16* bl = &xl[lr * XSTR];
        ah0[t].f4[0] = *(const float4*)(bh + kg * 8);
        ah0[t].f4[1] = *(const float4*)(bh + 16 + kg * 8);
        ah1[t].f4[0] = *(const float4*)(bh + 32 + kg * 8);
        ah1[t].f4[1] = *(const float4*)(bh + 48 + kg * 8);
        al0[t].f4[0] = *(const float4*)(bl + kg * 8);
        al0[t].f4[1] = *(const float4*)(bl + 16 + kg * 8);
        al1[t].f4[0] = *(const float4*)(bl + 32 + kg * 8);
        al1[t].f4[1] = *(const float4*)(bl + 48 + kg * 8);
    }

    float minv[2][8];
    int   mini[2][8];
#pragma unroll
    for (int t = 0; t < 2; ++t)
#pragma unroll
        for (int v = 0; v < 8; ++v) { minv[t][v] = 3.402823466e38f; mini[t][v] = 0; }

    for (int p = 0; p < NPHASE; ++p) {
#if HAS_TDM
        __builtin_amdgcn_s_wait_tensorcnt(0);
#endif
        __syncthreads();   // staged codebook phase visible to all waves

#pragma unroll 2
        for (int c = 0; c < KPHASE / 16; ++c) {
            const int krow = c * 16 + n;                  // row in staged buffer
            const __bf16* ph = &Wh_lds[krow * WSTR + kg * 8];
            const __bf16* pl = &Wl_lds[krow * WSTR + kg * 8];
            AFrag bh0, bh1, bl0, bl1;
#pragma unroll
            for (int j = 0; j < 4; ++j) {
                bh0.f2[j] = *(const float2*)(ph + 4 * (j & 1) + 16 * (j >> 1));
                bh1.f2[j] = *(const float2*)(ph + 32 + 4 * (j & 1) + 16 * (j >> 1));
                bl0.f2[j] = *(const float2*)(pl + 4 * (j & 1) + 16 * (j >> 1));
                bl1.f2[j] = *(const float2*)(pl + 32 + 4 * (j & 1) + 16 * (j >> 1));
            }
            const int   k  = p * KPHASE + c * 16 + n;     // this lane's column
            const float wn = wn_lds[k];
#pragma unroll
            for (int t = 0; t < 2; ++t) {
                v8f acc = {};
                acc = wmma_bf16(ah0[t], bh0, acc);
                acc = wmma_bf16(ah1[t], bh1, acc);
                acc = wmma_bf16(ah0[t], bl0, acc);
                acc = wmma_bf16(ah1[t], bl1, acc);
                acc = wmma_bf16(al0[t], bh0, acc);
                acc = wmma_bf16(al1[t], bh1, acc);
#pragma unroll
                for (int v = 0; v < 8; ++v) {             // C rows: M = v + 8*kg
                    float pr = wn - 2.f * acc[v];
                    if (pr < minv[t][v]) { minv[t][v] = pr; mini[t][v] = k; }
                }
            }
        }
        __syncthreads();   // all reads of staged buffer done
#if HAS_TDM
        if (p + 1 < NPHASE && wave == 0) {
            tdm_load_2d((unsigned)(size_t)(void*)Wh_lds,
                        (const char*)whi_u + (size_t)(p + 1) * KPHASE * DIM * 2, KPHASE);
            tdm_load_2d((unsigned)(size_t)(void*)Wl_lds,
                        (const char*)wlo_u + (size_t)(p + 1) * KPHASE * DIM * 2, KPHASE);
        }
#else
        if (p + 1 < NPHASE) {
            const size_t gb = (size_t)(p + 1) * KPHASE * DIM;
            for (int i = tid; i < KPHASE * DIM; i += 256) {
                int row = i >> 6, d = i & 63;
                Wh_lds[row * WSTR + d] = ((const __bf16*)whi_u)[gb + i];
                Wl_lds[row * WSTR + d] = ((const __bf16*)wlo_u)[gb + i];
            }
        }
#endif
    }

    // ---- (min, argmin) across the 16 column-lanes of each half ----
#pragma unroll
    for (int mask = 1; mask <= 8; mask <<= 1) {
#pragma unroll
        for (int t = 0; t < 2; ++t)
#pragma unroll
            for (int v = 0; v < 8; ++v) {
                float ov = __shfl_xor(minv[t][v], mask, 32);
                int   oi = __shfl_xor(mini[t][v], mask, 32);
                if (ov < minv[t][v] || (ov == minv[t][v] && oi < mini[t][v])) {
                    minv[t][v] = ov; mini[t][v] = oi;
                }
            }
    }
    if (n == 0) {
#pragma unroll
        for (int t = 0; t < 2; ++t)
#pragma unroll
            for (int v = 0; v < 8; ++v)
                ind_lds[wave * 32 + t * 16 + kg * 8 + v] = mini[t][v];
    }
    __syncthreads();

    // ---- gather f32 code rows to out[B,D,T], accumulate exact-f32 MSE ----
    float lsum = 0.f;
    const int r = tid;
    const int ind = ind_lds[r];
    for (int d = 0; d < DIM; ++d) {
        float q = W[(size_t)ind * DIM + d];
        float x = latBase[(size_t)d * TT + r];
        outBase[(size_t)d * TT + r] = q;
        float df = q - x;
        lsum += df * df;
    }
#pragma unroll
    for (int mask = 1; mask <= 16; mask <<= 1) lsum += __shfl_xor(lsum, mask, 32);
    if (lane == 0) red[wave] = lsum;
    __syncthreads();
    if (tid == 0) {
        float s = 0.f;
#pragma unroll
        for (int i = 0; i < 8; ++i) s += red[i];
        partials[blk] = s;
    }
}

// ---------------------------------------------------------------------------
// Kernel 2: deterministic finalize of vq_loss = (1+beta) * mean((q-x)^2)
// ---------------------------------------------------------------------------
__global__ __launch_bounds__(256) void vq_fin(const float* __restrict__ partials,
                                              float* __restrict__ loss) {
    __shared__ float sm[256];
    float s = 0.f;
    for (int i = threadIdx.x; i < NBLK; i += 256) s += partials[i];
    sm[threadIdx.x] = s;
    __syncthreads();
    for (int off = 128; off > 0; off >>= 1) {
        if (threadIdx.x < off) sm[threadIdx.x] += sm[threadIdx.x + off];
        __syncthreads();
    }
    if (threadIdx.x == 0)
        loss[0] = sm[0] * ((1.0f + BETA) / (float)(NROWS * DIM));
}

extern "C" void kernel_launch(void* const* d_in, const int* in_sizes, int n_in,
                              void* d_out, int out_size, void* d_ws, size_t ws_size,
                              hipStream_t stream) {
    (void)in_sizes; (void)n_in; (void)out_size; (void)ws_size;
    const float* lat = (const float*)d_in[0];   // [16, 64, 4096] f32
    const float* W   = (const float*)d_in[1];   // [1024, 64] f32
    float* out = (float*)d_out;                 // [16*64*4096] + 1 loss

    char* ws = (char*)d_ws;
    unsigned short* whi  = (unsigned short*)(ws);                    // 128 KB
    unsigned short* wlo  = (unsigned short*)(ws + 131072);           // 128 KB
    float*          wnrm = (float*)(ws + 262144);                    //   4 KB
    float*          part = (float*)(ws + 262144 + 4096);             //   1 KB

    vq_prep<<<(KCB + 255) / 256, 256, 0, stream>>>(W, whi, wlo, wnrm);
    vq_main<<<NBLK, 256, 0, stream>>>(lat, W, whi, wlo, wnrm, out, part);
    vq_fin<<<1, 256, 0, stream>>>(part, out + (size_t)NROWS * DIM);
}
